// KPConv_47081431499115
// MI455X (gfx1250) — matrix operand order repaired and checked by reading, not compiled, not run on hardware
//
#include <hip/hip_runtime.h>
#include <hip/hip_bf16.h>

// ---------------------------------------------------------------------------
// KPConv on MI455X (gfx1250, wave32)
//   phase 1: zero fp32 accumulator wf[N,480] in d_ws (96 MB, L2-resident)
//   phase 2: swizzle weights[15,32,64] -> bf16 WMMA B-fragments (61,440 B)
//   phase 3: edge scatter with global_atomic_add_f32 (1 wave / edge)
//   phase 4: out[N,64] = wf[N,480] x W via v_wmma_f32_16x16x32_bf16
// Workspace layout: [0, 96,000,000) fp32 wf ; [96,000,000, +61,440) bf16 Wswz
// ---------------------------------------------------------------------------

#define N_NODES 50000
#define N_EDGES 800000
#define KPTS 15
#define IN_DIM 32
#define OUT_DIM 64
#define KDIM (KPTS * IN_DIM)          // 480 = 15 chunks of 32
#define INV_EXTENT (1.0f / 0.6f)
#define WF_BYTES (N_NODES * KDIM * 4) // 96,000,000

typedef __attribute__((ext_vector_type(16))) __bf16 v16bf;
typedef __attribute__((ext_vector_type(8)))  float  v8f;

__device__ __forceinline__ unsigned short f2bf_bits(float x) {
  __bf16 h = (__bf16)x;
  unsigned short u;
  __builtin_memcpy(&u, &h, 2);
  return u;
}

// ---- phase 1: zero the accumulator (grid-stride, 128-bit stores) ----------
__global__ void kpconv_zero_wf(float4* __restrict__ p, int n4) {
  int i = blockIdx.x * blockDim.x + threadIdx.x;
  int stride = gridDim.x * blockDim.x;
  float4 z = make_float4(0.f, 0.f, 0.f, 0.f);
  for (; i < n4; i += stride) p[i] = z;
}

// ---- phase 2: weights -> bf16 B-fragment swizzle --------------------------
// Fragment f = c*4 + t covers B rows (k=c, i=0..31) x cols (o=t*16..t*16+15).
// 16-bit B 32x16 layout used here (contiguous K, not ISA table order): lane = half*16 + col; element j <-> K = half*16+j.
// Stored lane-major: wswz[(f*32 + lane)*16 + j], 32 B contiguous per lane.
__global__ void kpconv_swizzle_w(const float* __restrict__ w,
                                 unsigned short* __restrict__ wswz) {
  int frag = blockIdx.x;            // 0..59
  int lane = threadIdx.x;           // 0..31
  int c = frag >> 2, t = frag & 3;
  int col  = t * 16 + (lane & 15);
  int half = lane >> 4;
  unsigned short* dst = wswz + (frag * 32 + lane) * 16;
#pragma unroll
  for (int j = 0; j < 16; ++j) {
    int i = half * 16 + j;                       // input-dim row in chunk
    dst[j] = f2bf_bits(w[(c * IN_DIM + i) * OUT_DIM + col]);
  }
}

// ---- phase 3: edge scatter (1 wave per edge, lane = input feature) --------
__global__ void kpconv_edge_scatter(const float* __restrict__ pos,
                                    const float* __restrict__ feat,
                                    const float* __restrict__ kp,
                                    const int* __restrict__ esrc,
                                    const int* __restrict__ edst,
                                    float* __restrict__ wf) {
  int e = (blockIdx.x * blockDim.x + threadIdx.x) >> 5;  // wave-uniform
  if (e >= N_EDGES) return;
  int lane = threadIdx.x & 31;
  int s = esrc[e];
  int d = edst[e];
  float y0 = pos[s * 3 + 0] - pos[d * 3 + 0];
  float y1 = pos[s * 3 + 1] - pos[d * 3 + 1];
  float y2 = pos[s * 3 + 2] - pos[d * 3 + 2];
  float fi = feat[s * IN_DIM + lane];                    // coalesced 128 B
  float* base = wf + d * KDIM + lane;
#pragma unroll
  for (int k = 0; k < KPTS; ++k) {
    float dx = y0 - kp[k * 3 + 0];
    float dy = y1 - kp[k * 3 + 1];
    float dz = y2 - kp[k * 3 + 2];
    float dist = sqrtf(dx * dx + dy * dy + dz * dz);
    float m = 1.0f - dist * INV_EXTENT;
    if (m > 0.0f) {                                      // uniform branch
      float val = m * fi;
      float* addr = base + k * IN_DIM;
      // fp32 atomic executes in L2 (wf fits in 192 MB L2) — no HBM RMW
      asm volatile("global_atomic_add_f32 %0, %1, off"
                   :: "v"(addr), "v"(val) : "memory");
    }
  }
}

// ---- phase 4: WMMA GEMM  out[16-node tile, 64] --------------------------
// One wave per node tile (3125 tiles exactly -> EXEC always all-ones).
// A 16x32 bf16 layout: lane m = lane&15 is row; element j <-> K:
//   K = half*16 + j (matches B swizzle)
// -> two contiguous 32 B fp32 loads per lane, converted to bf16 in-register.
__global__ void __launch_bounds__(32)
kpconv_gemm(const float* __restrict__ wf,
            const unsigned short* __restrict__ wswz,
            float* __restrict__ out) {
  int tile = blockIdx.x;            // 0..3124
  int lane = threadIdx.x;           // 0..31
  int m    = lane & 15;
  int half = lane >> 4;
  int nodeBase = tile * 16;
  const float* row = wf + (nodeBase + m) * KDIM;

  v8f acc[4] = {v8f{}, v8f{}, v8f{}, v8f{}};

#pragma unroll
  for (int c = 0; c < KPTS; ++c) {
    const v8f f0 = *(const v8f*)(row + c * 32 + half * 16);
    const v8f f1 = *(const v8f*)(row + c * 32 + half * 16 + 8);
    v16bf a;
#pragma unroll
    for (int j = 0; j < 8; ++j) {
      a[j]     = (__bf16)f0[j];
      a[j + 8] = (__bf16)f1[j];
    }
#pragma unroll
    for (int t = 0; t < 4; ++t) {
      const v16bf b = *(const v16bf*)(wswz + ((c * 4 + t) * 32 + lane) * 16);
      acc[t] = __builtin_amdgcn_wmma_f32_16x16x32_bf16(
          /*neg_a=*/false, a, /*neg_b=*/false, b,
          /*c_mod=*/(short)0, acc[t], /*reuse_a=*/false, /*reuse_b=*/false);
    }
  }

  // D layout: VGPR r -> M = r (lanes 0-15) / r+8 (lanes 16-31); N = lane&15
#pragma unroll
  for (int t = 0; t < 4; ++t) {
#pragma unroll
    for (int r = 0; r < 8; ++r) {
      int node = nodeBase + r + half * 8;
      out[node * OUT_DIM + t * 16 + m] = acc[t][r];
    }
  }
}

extern "C" void kernel_launch(void* const* d_in, const int* in_sizes, int n_in,
                              void* d_out, int out_size, void* d_ws, size_t ws_size,
                              hipStream_t stream) {
  const float* pos  = (const float*)d_in[0];
  const float* feat = (const float*)d_in[1];
  const float* kp   = (const float*)d_in[2];
  const float* w    = (const float*)d_in[3];
  const int* esrc   = (const int*)d_in[4];
  const int* edst   = (const int*)d_in[5];
  float* out        = (float*)d_out;

  float* wf            = (float*)d_ws;                       // 96,000,000 B
  unsigned short* wswz = (unsigned short*)((char*)d_ws + WF_BYTES); // 61,440 B

  // 1) zero accumulator (harness never re-zeros d_ws between replays)
  int n4 = (N_NODES * KDIM) / 4;                             // 6,000,000
  kpconv_zero_wf<<<2048, 256, 0, stream>>>((float4*)wf, n4);

  // 2) weight swizzle: 60 fragments, 1 wave each
  kpconv_swizzle_w<<<60, 32, 0, stream>>>(w, wswz);

  // 3) edge scatter: 1 wave per edge, 8 edges per 256-thread block
  kpconv_edge_scatter<<<(N_EDGES + 7) / 8, 256, 0, stream>>>(
      pos, feat, kp, esrc, edst, wf);

  // 4) WMMA GEMM: 3125 node tiles, 1 wave each
  kpconv_gemm<<<N_NODES / 16, 32, 0, stream>>>(wf, wswz, out);
}
